// ParallelAttention_6150393168485
// MI455X (gfx1250) — matrix operand order repaired and checked by reading, not compiled, not run on hardware
//
#include <hip/hip_runtime.h>

typedef __bf16 bf16;
typedef __attribute__((ext_vector_type(16))) __bf16 bf16x16;
typedef __attribute__((ext_vector_type(8)))  float  f32x8;

constexpr int S_  = 2048;
constexpr int B_  = 2;
constexpr int H_  = 2048;
constexpr int NH_ = 16;
constexpr int DH_ = 128;
constexpr int TOK_ = S_ * B_;   // 4096 tokens, row t = s*B + b
constexpr int N3H_ = 3 * H_;    // 6144

constexpr int KB_   = 64;          // key block for attention
constexpr int KROW_ = DH_ + 8;     // padded K-tile row (bank-conflict-free)
constexpr int VROW_ = KB_ + 8;     // padded V-tile / P-tile row

__device__ __forceinline__ f32x8 zero8() {
  f32x8 z = {0.f, 0.f, 0.f, 0.f, 0.f, 0.f, 0.f, 0.f};
  return z;
}

// Load a 16-element bf16 fragment from two 16-byte-aligned 8-element groups.
__device__ __forceinline__ bf16x16 ldfrag(const bf16* p0, const bf16* p1) {
  union { bf16x16 v; uint4 q[2]; } u;
  u.q[0] = *reinterpret_cast<const uint4*>(p0);
  u.q[1] = *reinterpret_cast<const uint4*>(p1);
  return u.v;
}

__device__ __forceinline__ f32x8 wmma_bf16(bf16x16 a, bf16x16 b, f32x8 c) {
  // (neg_a, A, neg_b, B, c_mod, C, reuse_a, reuse_b)
  return __builtin_amdgcn_wmma_f32_16x16x32_bf16(false, a, false, b, (short)0, c,
                                                 false, false);
}

// Async DMA: global (16B per lane) -> LDS, tracked by ASYNCcnt, bypasses VGPRs.
// ldsoff = wave-relative LDS byte offset (low 32 bits of a __shared__ pointer,
// per the LDS aperture truncation rule).
__device__ __forceinline__ void async_g2l_b128(unsigned ldsoff, const void* gaddr) {
  asm volatile("global_load_async_to_lds_b128 %0, %1, off"
               :
               : "v"(ldsoff), "v"(gaddr)
               : "memory");
}

__device__ __forceinline__ void wait_async0() {
  asm volatile("s_wait_asynccnt 0x0" ::: "memory");
}

__global__ void cvt_f32_bf16(const float* __restrict__ s, bf16* __restrict__ d, int n) {
  int i = blockIdx.x * blockDim.x + threadIdx.x;
  if (i < n) d[i] = (bf16)s[i];
}

// ---------------------------------------------------------------------------
// QKV projection: C[4096 x 6144] = X(bf16) * Wqkv^T + bias, scattered into
// Q[b,n,s,d], K[b,n,s,d], Vt[b,n,d,s] (bf16).
// WG tile 128(M) x 128(N), 8 waves, each wave 32x64 (2x4 WMMA tiles).
// ---------------------------------------------------------------------------
__global__ __launch_bounds__(256) void qkv_gemm_kernel(
    const bf16* __restrict__ X, const bf16* __restrict__ W,
    const float* __restrict__ bias,
    bf16* __restrict__ Qg, bf16* __restrict__ Kg, bf16* __restrict__ Vt) {
  constexpr int K = H_;
  const int lane = threadIdx.x & 31;
  const int wave = threadIdx.x >> 5;
  const int hi = lane >> 4;
  const int ln = lane & 15;
  const int wm = wave >> 1;
  const int wn = wave & 1;
  const int m0 = blockIdx.x * 128 + wm * 32;
  const int n0 = blockIdx.y * 128 + wn * 64;

  f32x8 acc[2][4];
#pragma unroll
  for (int mi = 0; mi < 2; ++mi)
#pragma unroll
    for (int ni = 0; ni < 4; ++ni) acc[mi][ni] = zero8();

  for (int k0 = 0; k0 < K; k0 += 32) {
    bf16x16 af[2];
#pragma unroll
    for (int mi = 0; mi < 2; ++mi) {
      const bf16* r = X + (size_t)(m0 + mi * 16 + ln) * K + k0 + hi * 8;
      af[mi] = ldfrag(r, r + 16);
    }
    bf16x16 bfr[4];
#pragma unroll
    for (int ni = 0; ni < 4; ++ni) {
      const bf16* r = W + (size_t)(n0 + ni * 16 + ln) * K + k0 + hi * 8;
      bfr[ni] = ldfrag(r, r + 16);
    }
#pragma unroll
    for (int mi = 0; mi < 2; ++mi)
#pragma unroll
      for (int ni = 0; ni < 4; ++ni)
        acc[mi][ni] = wmma_bf16(af[mi], bfr[ni], acc[mi][ni]);
  }

  // Epilogue: C layout = lane holds col ln, rows r + hi*8 (r = VGPR index).
#pragma unroll
  for (int mi = 0; mi < 2; ++mi) {
#pragma unroll
    for (int ni = 0; ni < 4; ++ni) {
      const int col = n0 + ni * 16 + ln;          // column of mixed [*, 6144]
      const int head = col / (3 * DH_);
      const int rr = col - head * (3 * DH_);
      const int which = rr >> 7;                  // 0=q 1=k 2=v
      const int d = rr & (DH_ - 1);
      const float bv = bias[col];
#pragma unroll
      for (int r = 0; r < 8; ++r) {
        const int grow = m0 + mi * 16 + hi * 8 + r;   // token t = s*B + b
        const int s = grow / B_;
        const int bb = grow - s * B_;
        const float v = acc[mi][ni][r] + bv;
        const bf16 h = (bf16)v;
        const size_t hd = (size_t)(bb * NH_ + head);
        if (which == 0)      Qg[(hd * S_ + s) * DH_ + d] = h;
        else if (which == 1) Kg[(hd * S_ + s) * DH_ + d] = h;
        else                 Vt[(hd * DH_ + d) * S_ + s] = h;
      }
    }
  }
}

// ---------------------------------------------------------------------------
// Flash attention per (query block of 128, head). 8 waves x 16 query rows.
// K/V tiles staged in LDS per 64-key block via GLOBAL_LOAD_ASYNC_TO_LDS_B128
// (ASYNCcnt-tracked DMA, no VGPR round-trip); next block prefetched via
// global_prefetch_b8.
// ---------------------------------------------------------------------------
__global__ __launch_bounds__(256) void attn_kernel(
    const bf16* __restrict__ Qg, const bf16* __restrict__ Kg,
    const bf16* __restrict__ Vt, bf16* __restrict__ ctx) {
  __shared__ __align__(16) bf16 Kt[KB_][KROW_];        // 64 x 136  (17.4 KB)
  __shared__ __align__(16) bf16 Vts[DH_][VROW_];       // 128 x 72  (18.4 KB)
  __shared__ __align__(16) bf16 Ptile[8][16][VROW_];   // 8 x 16 x 72 (18.4 KB)

  const int tid = threadIdx.x;
  const int lane = tid & 31;
  const int wave = tid >> 5;
  const int hi = lane >> 4;
  const int ln = lane & 15;
  const int bh = blockIdx.y;            // b*NH + n
  const int bb = bh / NH_;
  const int nh = bh - bb * NH_;
  const int qb = blockIdx.x;
  const int q0 = qb * 128 + wave * 16;  // this wave's first query row

  const bf16* Qh = Qg + (size_t)bh * S_ * DH_;
  const bf16* Kh = Kg + (size_t)bh * S_ * DH_;
  const bf16* Vh = Vt + (size_t)bh * DH_ * S_;

  // Q fragments resident: 4 chunks of K=32 over DH=128.
  bf16x16 qf[4];
#pragma unroll
  for (int kc = 0; kc < 4; ++kc) {
    const bf16* r = Qh + (size_t)(q0 + ln) * DH_ + kc * 32 + hi * 8;
    qf[kc] = ldfrag(r, r + 16);
  }

  float m_i[8], l_i[8];
  f32x8 o[8];
#pragma unroll
  for (int r = 0; r < 8; ++r) { m_i[r] = -1e30f; l_i[r] = 0.f; }
#pragma unroll
  for (int nd = 0; nd < 8; ++nd) o[nd] = zero8();

  const float scale = 0.088388347648318447f;   // 1/sqrt(128)
  const int jmax = 2 * qb + 1;                 // causal: keys <= qb*128+127

  for (int j = 0; j <= jmax; ++j) {
    const int kbase = j * KB_;

    // --- Async-stage K tile (64 x 128 d) and V tile (128 d x 64 keys) ---
#pragma unroll
    for (int l = tid; l < KB_ * 16; l += 256) {          // 16 x b128 per K row
      const int row = l >> 4, c = (l & 15) * 8;
      async_g2l_b128((unsigned)(uintptr_t)&Kt[row][c],
                     Kh + (size_t)(kbase + row) * DH_ + c);
    }
#pragma unroll
    for (int l = tid; l < DH_ * 8; l += 256) {           // 8 x b128 per V row
      const int row = l >> 3, c = (l & 7) * 8;
      async_g2l_b128((unsigned)(uintptr_t)&Vts[row][c],
                     Vh + (size_t)row * S_ + kbase + c);
    }
    // Prefetch next key block (speculative; gfx1250 global_prefetch_b8).
    if (j < jmax) {
      if (tid < 128) {   // K: 64 rows x 256 B -> 128B-line pairs
        const int row = tid >> 1, half = tid & 1;
        __builtin_prefetch(Kh + (size_t)(kbase + KB_ + row) * DH_ + half * 64, 0, 3);
      } else {           // V: 128 rows x 128 B
        const int row = tid - 128;
        __builtin_prefetch(Vh + (size_t)row * S_ + kbase + KB_, 0, 3);
      }
    }
    wait_async0();       // each wave's DMA into LDS complete
    __syncthreads();     // visible workgroup-wide

    // --- scores S = Q K^T (fp32 acc), 4 key subtiles of 16 ---
    float sv[4][8];
#pragma unroll
    for (int nt = 0; nt < 4; ++nt) {
      f32x8 sacc = zero8();
#pragma unroll
      for (int kc = 0; kc < 4; ++kc) {
        const bf16* r = &Kt[nt * 16 + ln][kc * 32 + hi * 8];
        bf16x16 kf = ldfrag(r, r + 16);
        sacc = wmma_bf16(qf[kc], kf, sacc);
      }
      const int key = kbase + nt * 16 + ln;
#pragma unroll
      for (int r = 0; r < 8; ++r) {
        const int qrow = q0 + hi * 8 + r;
        float s = sacc[r] * scale;
        if (key > qrow) s = -10000.0f;         // masked_fill semantics
        sv[nt][r] = s;
      }
    }

    // --- online softmax; each row lives in one 16-lane half-group ---
#pragma unroll
    for (int r = 0; r < 8; ++r) {
      float mx = sv[0][r];
#pragma unroll
      for (int nt = 1; nt < 4; ++nt) mx = fmaxf(mx, sv[nt][r]);
      mx = fmaxf(mx, __shfl_xor(mx, 1, 32));
      mx = fmaxf(mx, __shfl_xor(mx, 2, 32));
      mx = fmaxf(mx, __shfl_xor(mx, 4, 32));
      mx = fmaxf(mx, __shfl_xor(mx, 8, 32));
      const float nm = fmaxf(m_i[r], mx);
      const float corr = __expf(m_i[r] - nm);
      float rs = 0.f;
#pragma unroll
      for (int nt = 0; nt < 4; ++nt) {
        const float p = __expf(sv[nt][r] - nm);
        sv[nt][r] = p;
        rs += p;
      }
      rs += __shfl_xor(rs, 1, 32);
      rs += __shfl_xor(rs, 2, 32);
      rs += __shfl_xor(rs, 4, 32);
      rs += __shfl_xor(rs, 8, 32);
      l_i[r] = l_i[r] * corr + rs;
      m_i[r] = nm;
#pragma unroll
      for (int nd = 0; nd < 8; ++nd) o[nd][r] *= corr;
    }

    // --- P: C-layout -> LDS -> A-layout fragments ---
#pragma unroll
    for (int nt = 0; nt < 4; ++nt)
#pragma unroll
      for (int r = 0; r < 8; ++r)
        Ptile[wave][hi * 8 + r][nt * 16 + ln] = (bf16)sv[nt][r];
    __syncthreads();

    // --- O += P * V  (2 k-chunks of 32 over the 64 keys) ---
#pragma unroll
    for (int kc = 0; kc < 2; ++kc) {
      const bf16* pr = &Ptile[wave][ln][kc * 32 + hi * 8];
      bf16x16 pf = ldfrag(pr, pr + 16);
#pragma unroll
      for (int nd = 0; nd < 8; ++nd) {
        const bf16* vr = &Vts[nd * 16 + ln][kc * 32 + hi * 8];
        bf16x16 vf = ldfrag(vr, vr + 16);
        o[nd] = wmma_bf16(pf, vf, o[nd]);
      }
    }
    __syncthreads();   // all LDS reads done before next block's staging
  }

  // Normalize + write ctx as [token(4096) x H] bf16 (token t = s*B + b).
#pragma unroll
  for (int nd = 0; nd < 8; ++nd) {
#pragma unroll
    for (int r = 0; r < 8; ++r) {
      const int srow = q0 + hi * 8 + r;
      const int col = nh * DH_ + nd * 16 + ln;
      const float v = o[nd][r] / l_i[r];
      ctx[((size_t)srow * B_ + bb) * H_ + col] = (bf16)v;
    }
  }
}

// ---------------------------------------------------------------------------
// Dense projection: out[4096 x 2048] (fp32) = ctx(bf16) * Wdense^T (skip bias).
// ---------------------------------------------------------------------------
__global__ __launch_bounds__(256) void dense_gemm_kernel(
    const bf16* __restrict__ A, const bf16* __restrict__ W,
    float* __restrict__ out) {
  constexpr int K = H_;
  constexpr int N = H_;
  const int lane = threadIdx.x & 31;
  const int wave = threadIdx.x >> 5;
  const int hi = lane >> 4;
  const int ln = lane & 15;
  const int wm = wave >> 1;
  const int wn = wave & 1;
  const int m0 = blockIdx.x * 128 + wm * 32;
  const int n0 = blockIdx.y * 128 + wn * 64;

  f32x8 acc[2][4];
#pragma unroll
  for (int mi = 0; mi < 2; ++mi)
#pragma unroll
    for (int ni = 0; ni < 4; ++ni) acc[mi][ni] = zero8();

  for (int k0 = 0; k0 < K; k0 += 32) {
    bf16x16 af[2];
#pragma unroll
    for (int mi = 0; mi < 2; ++mi) {
      const bf16* r = A + (size_t)(m0 + mi * 16 + ln) * K + k0 + hi * 8;
      af[mi] = ldfrag(r, r + 16);
    }
    bf16x16 bfr[4];
#pragma unroll
    for (int ni = 0; ni < 4; ++ni) {
      const bf16* r = W + (size_t)(n0 + ni * 16 + ln) * K + k0 + hi * 8;
      bfr[ni] = ldfrag(r, r + 16);
    }
#pragma unroll
    for (int mi = 0; mi < 2; ++mi)
#pragma unroll
      for (int ni = 0; ni < 4; ++ni)
        acc[mi][ni] = wmma_bf16(af[mi], bfr[ni], acc[mi][ni]);
  }

#pragma unroll
  for (int mi = 0; mi < 2; ++mi) {
#pragma unroll
    for (int ni = 0; ni < 4; ++ni) {
      const int col = n0 + ni * 16 + ln;
#pragma unroll
      for (int r = 0; r < 8; ++r) {
        const int grow = m0 + mi * 16 + hi * 8 + r;
        out[(size_t)grow * N + col] = acc[mi][ni][r];
      }
    }
  }
}

extern "C" void kernel_launch(void* const* d_in, const int* in_sizes, int n_in,
                              void* d_out, int out_size, void* d_ws, size_t ws_size,
                              hipStream_t stream) {
  const float* hidden  = (const float*)d_in[0];
  // d_in[1] = attention_mask: pure causal, reproduced analytically in-kernel.
  const float* w_qkv   = (const float*)d_in[2];
  const float* b_qkv   = (const float*)d_in[3];
  const float* w_dense = (const float*)d_in[4];
  float* out = (float*)d_out;

  char* ws = (char*)d_ws;
  auto take = [&](size_t elems) -> bf16* {
    bf16* p = (bf16*)ws;
    ws += ((elems * sizeof(bf16)) + 255) & ~(size_t)255;
    return p;
  };
  bf16* Xb  = take((size_t)TOK_ * H_);            // 16.8 MB
  bf16* Wq  = take((size_t)N3H_ * H_);            // 25.2 MB
  bf16* Wd  = take((size_t)H_ * H_);              //  8.4 MB
  bf16* Qg  = take((size_t)B_ * NH_ * S_ * DH_);  // 16.8 MB
  bf16* Kg  = take((size_t)B_ * NH_ * S_ * DH_);  // 16.8 MB
  bf16* Vt  = take((size_t)B_ * NH_ * S_ * DH_);  // 16.8 MB  (transposed V)
  bf16* Ctx = take((size_t)TOK_ * H_);            // 16.8 MB

  {
    int n = TOK_ * H_;
    cvt_f32_bf16<<<(n + 255) / 256, 256, 0, stream>>>(hidden, Xb, n);
  }
  {
    int n = N3H_ * H_;
    cvt_f32_bf16<<<(n + 255) / 256, 256, 0, stream>>>(w_qkv, Wq, n);
  }
  {
    int n = H_ * H_;
    cvt_f32_bf16<<<(n + 255) / 256, 256, 0, stream>>>(w_dense, Wd, n);
  }

  qkv_gemm_kernel<<<dim3(TOK_ / 128, N3H_ / 128), 256, 0, stream>>>(
      Xb, Wq, b_qkv, Qg, Kg, Vt);
  attn_kernel<<<dim3(S_ / 128, B_ * NH_), 256, 0, stream>>>(Qg, Kg, Vt, Ctx);
  dense_gemm_kernel<<<dim3(TOK_ / 128, H_ / 128), 256, 0, stream>>>(Ctx, Wd, out);

  (void)in_sizes; (void)n_in; (void)out_size; (void)ws_size;
}
